// Attention_18545668784927
// MI455X (gfx1250) — matrix operand order repaired and checked by reading, not compiled, not run on hardware
//
#include <hip/hip_runtime.h>

// ---------------- problem constants ----------------
#define DIMX   1024
#define BATCH  4
#define SEQ    2048
#define NHEAD  8
#define HDIM   128
#define WINDOW 16
#define MROWS  (BATCH * SEQ)     // 8192
#define EPSV   1e-6f

// ---------------- WMMA fragment types ----------------
typedef __attribute__((ext_vector_type(16))) __bf16        v16bf;
typedef __attribute__((ext_vector_type(8)))  float         v8f;
typedef __attribute__((ext_vector_type(4)))  unsigned int  v4u;

union Frag {
  v4u            q[2];   // two 16B loads
  v16bf          v;      // WMMA A/B operand
  unsigned short us[16]; // element poke for gathers
};

__device__ __forceinline__ unsigned short f32_to_bf16(float f) {
  unsigned int u = __float_as_uint(f);
  unsigned int r = 0x7FFFu + ((u >> 16) & 1u);   // round-to-nearest-even
  return (unsigned short)((u + r) >> 16);
}

// ---------------- workspace layout (bf16 elements) ----------------
#define OFF_XB 0
#define OFF_WQ (OFF_XB + MROWS * DIMX)
#define OFF_WK (OFF_WQ + DIMX * DIMX)
#define OFF_WV (OFF_WK + DIMX * DIMX)
#define OFF_WO (OFF_WV + DIMX * DIMX)
#define OFF_QB (OFF_WO + DIMX * DIMX)
#define OFF_KB (OFF_QB + MROWS * DIMX)
#define OFF_VB (OFF_KB + MROWS * DIMX)
#define OFF_AB (OFF_VB + MROWS * DIMX)
// total = 46,137,344 bf16 elems ~= 92 MB

// ============================================================
// 1) f32 -> bf16 conversion, 8 elements / thread-iteration
//    (pure HBM-bandwidth kernel: 2x b128 load -> 1x b128 store)
// ============================================================
__global__ void __launch_bounds__(256)
cvt_kernel(const float* __restrict__ x,
           const float* __restrict__ wq, const float* __restrict__ wk,
           const float* __restrict__ wv, const float* __restrict__ wo,
           unsigned short* __restrict__ ws) {
  const long long NX = (long long)MROWS * DIMX;
  const long long NW = (long long)DIMX * DIMX;
  const long long ngroups = (NX + 4 * NW) / 8;   // all sizes divisible by 8
  const long long NXG = NX / 8, NWG = NW / 8;
  const long long stride = (long long)gridDim.x * blockDim.x;
  for (long long g = (long long)blockIdx.x * blockDim.x + threadIdx.x;
       g < ngroups; g += stride) {
    const float* src;
    long long k, o;
    if (g < NXG) {
      src = x; k = g * 8; o = OFF_XB + k;
    } else {
      long long j = g - NXG;
      int w = (int)(j / NWG);
      k = (j - (long long)w * NWG) * 8;
      src = (w == 0) ? wq : (w == 1) ? wk : (w == 2) ? wv : wo;
      o = OFF_WQ + (long long)w * NW + k;
    }
    union { v4u q; float f[4]; } a, b;
    a.q = *reinterpret_cast<const v4u*>(src + k);
    b.q = *reinterpret_cast<const v4u*>(src + k + 4);
    union { v4u q; unsigned short us[8]; } r;
#pragma unroll
    for (int j = 0; j < 4; j++) r.us[j]     = f32_to_bf16(a.f[j]);
#pragma unroll
    for (int j = 0; j < 4; j++) r.us[4 + j] = f32_to_bf16(b.f[j]);
    *reinterpret_cast<v4u*>(ws + o) = r.q;
  }
}

// ============================================================
// 2) QKV projections with fused RMSNorm (Q,K) — bf16 WMMA
//    grid.x = M/32, grid.y = 3 (Q,K,V); 8 waves/block.
//    Each wave: 32 M-rows x 128 N-cols (2 A-frags, 8 B-tiles,
//    16 accumulators) -> each B fragment feeds 2 WMMAs.
// ============================================================
__global__ void __launch_bounds__(256)
qkv_gemm_kernel(unsigned short* __restrict__ ws,
                const float* __restrict__ qnw,
                const float* __restrict__ knw) {
  const int mat   = blockIdx.y;          // 0=Q 1=K 2=V
  const int mbase = blockIdx.x * 32;
  const int lane  = threadIdx.x & 31;
  const int wave  = threadIdx.x >> 5;
  const int ln15  = lane & 15;
  const int hi    = lane >> 4;
  const int wbase = wave * 128;

  const unsigned short* xb = ws + OFF_XB;
  const unsigned short* W  = ws + OFF_WQ + (size_t)mat * (DIMX * DIMX);
  unsigned short* dst = ws + ((mat == 0) ? OFF_QB : (mat == 1) ? OFF_KB : OFF_VB);

  v8f accA[8] = {};
  v8f accB[8] = {};
  const size_t arow0 = (size_t)(mbase + ln15) * DIMX;
  const size_t arow1 = arow0 + (size_t)16 * DIMX;

  for (int kc = 0; kc < DIMX; kc += 32) {
    // WGP-scope prefetch of next K-chunk of the A rows (global_prefetch_b8)
    __builtin_prefetch(xb + arow0 + kc + 32, 0, 3);
    __builtin_prefetch(xb + arow1 + kc + 32, 0, 3);
    Frag a0, a1;
    a0.q[0] = *reinterpret_cast<const v4u*>(xb + arow0 + kc + hi * 8);
    a0.q[1] = *reinterpret_cast<const v4u*>(xb + arow0 + kc + 16 + hi * 8);
    a1.q[0] = *reinterpret_cast<const v4u*>(xb + arow1 + kc + hi * 8);
    a1.q[1] = *reinterpret_cast<const v4u*>(xb + arow1 + kc + 16 + hi * 8);
#pragma unroll
    for (int t = 0; t < 8; t++) {
      Frag bf;
      const unsigned short* wp =
          W + (size_t)(wbase + t * 16 + ln15) * DIMX + kc + hi * 8;
      bf.q[0] = *reinterpret_cast<const v4u*>(wp);
      bf.q[1] = *reinterpret_cast<const v4u*>(wp + 16);
      accA[t] = __builtin_amdgcn_wmma_f32_16x16x32_bf16(
          false, a0.v, false, bf.v, (short)0, accA[t], false, false);
      accB[t] = __builtin_amdgcn_wmma_f32_16x16x32_bf16(
          false, a1.v, false, bf.v, (short)0, accB[t], false, false);
    }
  }

  if (mat < 2) {
    // fused RMSNorm over the full 1024-wide rows (32 rows)
    const float* nw = (mat == 0) ? qnw : knw;
    float lsA[8], lsB[8];
#pragma unroll
    for (int i = 0; i < 8; i++) {
      float sa = 0.f, sb = 0.f;
#pragma unroll
      for (int t = 0; t < 8; t++) {
        sa += accA[t][i] * accA[t][i];
        sb += accB[t][i] * accB[t][i];
      }
      lsA[i] = sa; lsB[i] = sb;
    }
    // reduce across the 16 lanes sharing the same rows (masks <16 stay in half)
#pragma unroll
    for (int msk = 1; msk < 16; msk <<= 1)
#pragma unroll
      for (int i = 0; i < 8; i++) {
        lsA[i] += __shfl_xor(lsA[i], msk, 32);
        lsB[i] += __shfl_xor(lsB[i], msk, 32);
      }

    __shared__ float red[32][8];
    if (ln15 == 0) {
#pragma unroll
      for (int i = 0; i < 8; i++) red[hi * 8 + i][wave]      = lsA[i];
#pragma unroll
      for (int i = 0; i < 8; i++) red[16 + hi * 8 + i][wave] = lsB[i];
    }
    __syncthreads();

    float scaleA[8], scaleB[8];
#pragma unroll
    for (int i = 0; i < 8; i++) {
      float ta = 0.f, tbv = 0.f;
#pragma unroll
      for (int w = 0; w < 8; w++) {
        ta  += red[hi * 8 + i][w];
        tbv += red[16 + hi * 8 + i][w];
      }
      scaleA[i] = rsqrtf(ta  * (1.0f / DIMX) + EPSV);
      scaleB[i] = rsqrtf(tbv * (1.0f / DIMX) + EPSV);
    }
#pragma unroll
    for (int t = 0; t < 8; t++) {
      int col = wbase + t * 16 + ln15;
      float g = nw[col];
#pragma unroll
      for (int i = 0; i < 8; i++) {
        int rowA = mbase + hi * 8 + i;
        int rowB = rowA + 16;
        dst[(size_t)rowA * DIMX + col] = f32_to_bf16(accA[t][i] * scaleA[i] * g);
        dst[(size_t)rowB * DIMX + col] = f32_to_bf16(accB[t][i] * scaleB[i] * g);
      }
    }
  } else {
#pragma unroll
    for (int t = 0; t < 8; t++) {
      int col = wbase + t * 16 + ln15;
#pragma unroll
      for (int i = 0; i < 8; i++) {
        int rowA = mbase + hi * 8 + i;
        int rowB = rowA + 16;
        dst[(size_t)rowA * DIMX + col] = f32_to_bf16(accA[t][i]);
        dst[(size_t)rowB * DIMX + col] = f32_to_bf16(accB[t][i]);
      }
    }
  }
}

// ============================================================
// 3) Sliding-window attention: one wave per (b, h, 16-query tile).
//    Keys span two 16-blocks: [qbase-16, qbase) and [qbase, qbase+16).
// ============================================================
__global__ void __launch_bounds__(32)
attn_kernel(unsigned short* __restrict__ ws) {
  const int qt   = blockIdx.x;
  const int h    = blockIdx.y;
  const int b    = blockIdx.z;
  const int lane = threadIdx.x & 31;
  const int ln15 = lane & 15;
  const int hi   = lane >> 4;
  const int qbase = qt * 16;
  const size_t tb = (size_t)b * SEQ;
  const int hoff  = h * HDIM;

  const unsigned short* qb = ws + OFF_QB;
  const unsigned short* kb = ws + OFF_KB;
  const unsigned short* vb = ws + OFF_VB;
  unsigned short* ab = ws + OFF_AB;

  // ---- scores: Q(16x128) . K^T over two key blocks ----
  v8f s0 = {}, s1 = {};
  const size_t qrow = (tb + qbase + ln15) * DIMX + hoff;
  int kt0 = qbase - 16 + ln15; if (kt0 < 0) kt0 = 0;  // clamped, masked later
  int kt1 = qbase + ln15;
  const size_t krow0 = (tb + kt0) * DIMX + hoff;
  const size_t krow1 = (tb + kt1) * DIMX + hoff;

#pragma unroll
  for (int dc = 0; dc < HDIM; dc += 32) {
    Frag a, b0, b1;
    a.q[0]  = *reinterpret_cast<const v4u*>(qb + qrow + dc + hi * 8);
    a.q[1]  = *reinterpret_cast<const v4u*>(qb + qrow + dc + 16 + hi * 8);
    b0.q[0] = *reinterpret_cast<const v4u*>(kb + krow0 + dc + hi * 8);
    b0.q[1] = *reinterpret_cast<const v4u*>(kb + krow0 + dc + 16 + hi * 8);
    b1.q[0] = *reinterpret_cast<const v4u*>(kb + krow1 + dc + hi * 8);
    b1.q[1] = *reinterpret_cast<const v4u*>(kb + krow1 + dc + 16 + hi * 8);
    s0 = __builtin_amdgcn_wmma_f32_16x16x32_bf16(false, a.v, false, b0.v,
                                                 (short)0, s0, false, false);
    s1 = __builtin_amdgcn_wmma_f32_16x16x32_bf16(false, a.v, false, b1.v,
                                                 (short)0, s1, false, false);
  }

  // ---- ALiBi bias + causal/window mask + softmax ----
  const float slope = 1.0f / (float)(1 << (h + 1));   // 2^-(h+1)
  const float inv_sqrt_d = 0.0883883476f;             // 1/sqrt(128)
  const int j0 = qbase - 16 + ln15;
  const int j1 = qbase + ln15;
  float p0[8], p1[8], m[8];
#pragma unroll
  for (int i = 0; i < 8; i++) {
    int qi = qbase + hi * 8 + i;
    int r0 = j0 - qi, r1 = j1 - qi;
    float v0 = (j0 >= 0 && r0 <= 0 && r0 >= -WINDOW)
                   ? s0[i] * inv_sqrt_d + slope * (float)r0 : -1e30f;
    float v1 = (r1 <= 0 && r1 >= -WINDOW)
                   ? s1[i] * inv_sqrt_d + slope * (float)r1 : -1e30f;
    p0[i] = v0; p1[i] = v1;
    m[i] = fmaxf(v0, v1);
  }
#pragma unroll
  for (int msk = 1; msk < 16; msk <<= 1)
#pragma unroll
    for (int i = 0; i < 8; i++) m[i] = fmaxf(m[i], __shfl_xor(m[i], msk, 32));
  float sum[8];
#pragma unroll
  for (int i = 0; i < 8; i++) {
    p0[i] = __expf(p0[i] - m[i]);
    p1[i] = __expf(p1[i] - m[i]);
    sum[i] = p0[i] + p1[i];
  }
#pragma unroll
  for (int msk = 1; msk < 16; msk <<= 1)
#pragma unroll
    for (int i = 0; i < 8; i++) sum[i] += __shfl_xor(sum[i], msk, 32);

  // ---- transpose P (C-layout -> A-fragment layout) through LDS ----
  __shared__ unsigned short plds[16][32];
#pragma unroll
  for (int i = 0; i < 8; i++) {
    float inv = 1.0f / sum[i];   // >0: self key always valid
    plds[hi * 8 + i][ln15]      = f32_to_bf16(p0[i] * inv);
    plds[hi * 8 + i][16 + ln15] = f32_to_bf16(p1[i] * inv);
  }
  __syncthreads();

  Frag pa;
  {
    const int mrow = ln15;
    const int ks = hi * 8;
#pragma unroll
    for (int j = 0; j < 8; j++) pa.us[j]     = plds[mrow][ks + j];
#pragma unroll
    for (int j = 0; j < 8; j++) pa.us[8 + j] = plds[mrow][16 + ks + j];
  }

  // ---- out = P(16x32) . V(32x128) ----
#pragma unroll
  for (int t = 0; t < 8; t++) {
    Frag vf;
    const int dcol = hoff + t * 16 + ln15;
#pragma unroll
    for (int j = 0; j < 16; j++) {
      int k = (j < 8) ? (hi * 8 + j) : (16 + hi * 8 + (j - 8));
      int tok = qbase - 16 + k; if (tok < 0) tok = 0;  // masked rows have p=0
      vf.us[j] = vb[(tb + tok) * DIMX + dcol];
    }
    v8f o = {};
    o = __builtin_amdgcn_wmma_f32_16x16x32_bf16(false, pa.v, false, vf.v,
                                                (short)0, o, false, false);
#pragma unroll
    for (int i = 0; i < 8; i++) {
      int row = qbase + hi * 8 + i;
      ab[(tb + row) * DIMX + dcol] = f32_to_bf16(o[i]);
    }
  }
}

// ============================================================
// 4) Output projection: attn_out(bf16) @ wo^T -> f32 out
//    Same 32-row x 128-col wave tiling as the QKV GEMM.
// ============================================================
__global__ void __launch_bounds__(256)
out_gemm_kernel(const unsigned short* __restrict__ ws, float* __restrict__ out) {
  const int mbase = blockIdx.x * 32;
  const int lane  = threadIdx.x & 31;
  const int wave  = threadIdx.x >> 5;
  const int ln15  = lane & 15;
  const int hi    = lane >> 4;
  const int wbase = wave * 128;

  const unsigned short* ain = ws + OFF_AB;
  const unsigned short* W   = ws + OFF_WO;

  v8f accA[8] = {};
  v8f accB[8] = {};
  const size_t arow0 = (size_t)(mbase + ln15) * DIMX;
  const size_t arow1 = arow0 + (size_t)16 * DIMX;

  for (int kc = 0; kc < DIMX; kc += 32) {
    __builtin_prefetch(ain + arow0 + kc + 32, 0, 3);
    __builtin_prefetch(ain + arow1 + kc + 32, 0, 3);
    Frag a0, a1;
    a0.q[0] = *reinterpret_cast<const v4u*>(ain + arow0 + kc + hi * 8);
    a0.q[1] = *reinterpret_cast<const v4u*>(ain + arow0 + kc + 16 + hi * 8);
    a1.q[0] = *reinterpret_cast<const v4u*>(ain + arow1 + kc + hi * 8);
    a1.q[1] = *reinterpret_cast<const v4u*>(ain + arow1 + kc + 16 + hi * 8);
#pragma unroll
    for (int t = 0; t < 8; t++) {
      Frag bf;
      const unsigned short* wp =
          W + (size_t)(wbase + t * 16 + ln15) * DIMX + kc + hi * 8;
      bf.q[0] = *reinterpret_cast<const v4u*>(wp);
      bf.q[1] = *reinterpret_cast<const v4u*>(wp + 16);
      accA[t] = __builtin_amdgcn_wmma_f32_16x16x32_bf16(
          false, a0.v, false, bf.v, (short)0, accA[t], false, false);
      accB[t] = __builtin_amdgcn_wmma_f32_16x16x32_bf16(
          false, a1.v, false, bf.v, (short)0, accB[t], false, false);
    }
  }

#pragma unroll
  for (int t = 0; t < 8; t++) {
    int col = wbase + t * 16 + ln15;
#pragma unroll
    for (int i = 0; i < 8; i++) {
      int rowA = mbase + hi * 8 + i;
      int rowB = rowA + 16;
      out[(size_t)rowA * DIMX + col] = accA[t][i];
      out[(size_t)rowB * DIMX + col] = accB[t][i];
    }
  }
}

// ============================================================
extern "C" void kernel_launch(void* const* d_in, const int* in_sizes, int n_in,
                              void* d_out, int out_size, void* d_ws, size_t ws_size,
                              hipStream_t stream) {
  (void)in_sizes; (void)n_in; (void)out_size; (void)ws_size;
  const float* x   = (const float*)d_in[0];
  const float* wq  = (const float*)d_in[1];
  const float* wk  = (const float*)d_in[2];
  const float* wv  = (const float*)d_in[3];
  const float* wo  = (const float*)d_in[4];
  const float* qnw = (const float*)d_in[5];
  const float* knw = (const float*)d_in[6];
  unsigned short* ws = (unsigned short*)d_ws;
  float* out = (float*)d_out;

  cvt_kernel<<<1024, 256, 0, stream>>>(x, wq, wk, wv, wo, ws);
  qkv_gemm_kernel<<<dim3(MROWS / 32, 3), 256, 0, stream>>>(ws, qnw, knw);
  attn_kernel<<<dim3(SEQ / 16, NHEAD, BATCH), 32, 0, stream>>>(ws);
  out_gemm_kernel<<<dim3(MROWS / 32), 256, 0, stream>>>(ws, out);
}
